// IPAdapterAttnProcessor2_0_11596411699932
// MI455X (gfx1250) — matrix operand order repaired and checked by reading, not compiled, not run on hardware
//
#include <hip/hip_runtime.h>

typedef __attribute__((ext_vector_type(16))) __bf16 v16bf;
typedef __attribute__((ext_vector_type(8)))  __bf16 v8bf;
typedef __attribute__((ext_vector_type(8)))  float  v8f;

#define B_   2
#define L_   4096
#define S_   77
#define SP_  96          // S padded to 3 K-steps of 32
#define TIP  16
#define TP_  32          // T_ip padded to 1 K-step of 32
#define C_   1280
#define CC_  768
#define H_   20
#define DH_  64

__device__ __forceinline__ __bf16 f2bf(float f) { return (__bf16)f; }

__device__ __forceinline__ v16bf mk16(uint4 lo, uint4 hi) {
    union { uint4 q[2]; v16bf v; } u;
    u.q[0] = lo; u.q[1] = hi;
    return u.v;
}
// A-fragment: 16x32 bf16 A-layout. Lane (m = lane&15, hf = lane>>4) holds
// k = {8hf..8hf+7, 16+8hf..16+8hf+7}: two 16B loads from row base p.
__device__ __forceinline__ v16bf loadA(const __bf16* p, int hf) {
    uint4 lo = *(const uint4*)(p + (hf << 3));
    uint4 hi = *(const uint4*)(p + 16 + (hf << 3));
    return mk16(lo, hi);
}
// B-fragment: 32x16 bf16 B-layout. Lane (n = lane&15, hf) holds
// k = {8hf..8hf+7, 16+8hf..16+8hf+7}: two 16B loads from a k-major column base p.
__device__ __forceinline__ v16bf loadB(const __bf16* p, int hf) {
    uint4 lo = *(const uint4*)(p + (hf << 3));
    uint4 hi = *(const uint4*)(p + 16 + (hf << 3));
    return mk16(lo, hi);
}

// ---------------------------------------------------------------------------
// Prep 1: elementwise f32 -> bf16 (8 elems/thread, vector load+store)
// ---------------------------------------------------------------------------
__global__ __launch_bounds__(256) void cvt_bf(const float* __restrict__ in,
                                              __bf16* __restrict__ out, int n8) {
    int i = blockIdx.x * 256 + threadIdx.x;
    if (i >= n8) return;
    float4 f0 = ((const float4*)in)[2 * i];
    float4 f1 = ((const float4*)in)[2 * i + 1];
    v8bf v;
    v[0] = f2bf(f0.x); v[1] = f2bf(f0.y); v[2] = f2bf(f0.z); v[3] = f2bf(f0.w);
    v[4] = f2bf(f1.x); v[5] = f2bf(f1.y); v[6] = f2bf(f1.z); v[7] = f2bf(f1.w);
    ((v8bf*)out)[i] = v;
}

// ---------------------------------------------------------------------------
// Prep 2: W[K,N] f32 -> WT[N,K] bf16 (32x32 LDS tile transpose, coalesced)
// ---------------------------------------------------------------------------
__global__ __launch_bounds__(256) void transcvt(const float* __restrict__ W,
                                                __bf16* __restrict__ WT,
                                                int K, int N) {
    __shared__ float t[32][33];
    const int tiles_k = K >> 5;
    const int k0 = (blockIdx.x % tiles_k) << 5;
    const int n0 = (blockIdx.x / tiles_k) << 5;
    const int tx = threadIdx.x & 31;
    const int ty = threadIdx.x >> 5;
    #pragma unroll
    for (int r = 0; r < 4; ++r)
        t[ty + 8 * r][tx] = W[(size_t)(k0 + ty + 8 * r) * N + n0 + tx];
    __syncthreads();
    #pragma unroll
    for (int r = 0; r < 4; ++r)
        WT[(size_t)(n0 + ty + 8 * r) * K + k0 + tx] = f2bf(t[tx][ty + 8 * r]);
}

// ---------------------------------------------------------------------------
// Register-blocked, double-buffered GEMM: each wave computes a 64x32 output
// block (4 M-subtiles x 2 N-subtiles). K advances by 64 with two fragment
// buffers so the 12 b128 loads of one phase are in flight during the other
// phase's 8 independent v_wmma_f32_16x16x32_bf16.
//   out_bf16[M,N] = A_bf16[M,K] @ WT_bf16[N,K]^T   (K % 64 == 0)
// ---------------------------------------------------------------------------
__global__ __launch_bounds__(256) void proj_gemm_blk(
    const __bf16* __restrict__ A, const __bf16* __restrict__ WT,
    __bf16* __restrict__ out, int M, int N, int K, int tiles_n32)
{
    const int wave = threadIdx.x >> 5;
    const int lane = threadIdx.x & 31;
    const int wid  = blockIdx.x * 8 + wave;
    const int tm   = wid / tiles_n32;
    const int tn   = wid % tiles_n32;
    const int hf   = lane >> 4;
    const int nl   = lane & 15;

    const __bf16* ap[4];
    #pragma unroll
    for (int i = 0; i < 4; ++i) {
        int r = tm * 64 + i * 16 + nl;
        if (r >= M) r = M - 1;                     // clamp; stores guarded
        ap[i] = A + (size_t)r * K;
    }
    const __bf16* bp[2];
    #pragma unroll
    for (int j = 0; j < 2; ++j)
        bp[j] = WT + (size_t)(tn * 32 + j * 16 + nl) * K;

    v8f acc[4][2] = {};
    v16bf a0[4], b0[2], a1[4], b1[2];
    #pragma unroll
    for (int i = 0; i < 4; ++i) a0[i] = loadA(ap[i], hf);
    #pragma unroll
    for (int j = 0; j < 2; ++j) b0[j] = loadB(bp[j], hf);

    for (int k0 = 0; k0 < K; k0 += 64) {
        // prefetch phase-1 fragments (k0+32) while phase-0 WMMAs run
        #pragma unroll
        for (int i = 0; i < 4; ++i) a1[i] = loadA(ap[i] + k0 + 32, hf);
        #pragma unroll
        for (int j = 0; j < 2; ++j) b1[j] = loadB(bp[j] + k0 + 32, hf);
        #pragma unroll
        for (int i = 0; i < 4; ++i)
            #pragma unroll
            for (int j = 0; j < 2; ++j)
                acc[i][j] = __builtin_amdgcn_wmma_f32_16x16x32_bf16(
                    false, a0[i], false, b0[j], (short)0, acc[i][j], false, false);
        // prefetch next phase-0 fragments (k0+64) while phase-1 WMMAs run
        if (k0 + 64 < K) {
            #pragma unroll
            for (int i = 0; i < 4; ++i) a0[i] = loadA(ap[i] + k0 + 64, hf);
            #pragma unroll
            for (int j = 0; j < 2; ++j) b0[j] = loadB(bp[j] + k0 + 64, hf);
        }
        #pragma unroll
        for (int i = 0; i < 4; ++i)
            #pragma unroll
            for (int j = 0; j < 2; ++j)
                acc[i][j] = __builtin_amdgcn_wmma_f32_16x16x32_bf16(
                    false, a1[i], false, b1[j], (short)0, acc[i][j], false, false);
    }

    if (tm * 64 + 64 <= M) {                       // full-tile fast path
        #pragma unroll
        for (int i = 0; i < 4; ++i)
            #pragma unroll
            for (int j = 0; j < 2; ++j) {
                const int ncol = tn * 32 + j * 16 + nl;
                #pragma unroll
                for (int r = 0; r < 8; ++r)
                    out[(size_t)(tm * 64 + i * 16 + r + (hf << 3)) * N + ncol] =
                        f2bf(acc[i][j][r]);
            }
    } else {
        #pragma unroll
        for (int i = 0; i < 4; ++i)
            #pragma unroll
            for (int j = 0; j < 2; ++j) {
                const int ncol = tn * 32 + j * 16 + nl;
                #pragma unroll
                for (int r = 0; r < 8; ++r) {
                    const int m = tm * 64 + i * 16 + r + (hf << 3);
                    if (m < M) out[(size_t)m * N + ncol] = f2bf(acc[i][j][r]);
                }
            }
    }
}

// ---------------------------------------------------------------------------
// V projection (single 16x16 tile per wave, tiny M): store TRANSPOSED and
// row-padded: outT[b][c][spad] so the PV B-fragment is k-contiguous.
// ---------------------------------------------------------------------------
__global__ __launch_bounds__(256) void proj_gemm_T(
    const __bf16* __restrict__ A, const __bf16* __restrict__ WT,
    __bf16* __restrict__ outT, int M, int N, int K, int tiles_n,
    int Sdim, int SPad)
{
    const int wave = threadIdx.x >> 5;
    const int lane = threadIdx.x & 31;
    const int wid  = blockIdx.x * 8 + wave;
    const int tm   = wid / tiles_n;
    const int tn   = wid % tiles_n;
    const int hf   = lane >> 4;
    const int nl   = lane & 15;

    int arow = tm * 16 + nl;
    if (arow >= M) arow = M - 1;
    const int ncol = tn * 16 + nl;

    const __bf16* ap = A  + (size_t)arow * K;
    const __bf16* bp = WT + (size_t)ncol * K;
    v8f acc = {};
    for (int k0 = 0; k0 < K; k0 += 32) {
        v16bf a = loadA(ap + k0, hf);
        v16bf b = loadB(bp + k0, hf);
        acc = __builtin_amdgcn_wmma_f32_16x16x32_bf16(
            false, a, false, b, (short)0, acc, false, false);
    }
    #pragma unroll
    for (int r = 0; r < 8; ++r) {
        const int m = tm * 16 + r + (hf << 3);
        if (m < M) {
            const int bb = m / Sdim, s = m % Sdim;
            outT[((size_t)bb * N + ncol) * SPad + s] = f2bf(acc[r]);
        }
    }
}

// ---------------------------------------------------------------------------
// Fused biased-softmax cross-attention + IP stream. P staged in LDS as bf16.
// ---------------------------------------------------------------------------
#define PIP_  1536                      // bf16 elems: 16x96 P, then 16x32 P_ip

__global__ __launch_bounds__(256) void attn_fused(
    const __bf16* __restrict__ Qbf, const __bf16* __restrict__ Kbf,
    const __bf16* __restrict__ VT,  const __bf16* __restrict__ Kip,
    const __bf16* __restrict__ VipT, const float* __restrict__ region,
    const float* __restrict__ sigma_p, __bf16* __restrict__ attn_out)
{
    __shared__ __bf16 smem[8][PIP_ + 16 * TP_];   // 4KB/wave, 32KB total

    const int wave = threadIdx.x >> 5;
    const int lane = threadIdx.x & 31;
    const int wid  = blockIdx.x * 8 + wave;
    const int b    = wid / (H_ * 256);
    const int rem  = wid % (H_ * 256);
    const int h    = rem / 256;
    const int lt   = rem % 256;
    const int hf   = lane >> 4;
    const int nl   = lane & 15;

    const float scale = 0.125f;                   // 1/sqrt(64)
    const float sigw  = sigma_p[0] * 0.3f;        // sigma * REGION_WEIGHT

    // Q fragments (A layout): DH=64 -> two K=32 steps, two b128 loads each
    const __bf16* qp = Qbf + (size_t)(b * L_ + lt * 16 + nl) * C_ + h * DH_;
    v16bf aQ[2];
    #pragma unroll
    for (int kt = 0; kt < 2; ++kt) aQ[kt] = loadA(qp + kt * 32, hf);

    // scores = Q K^T : 5 S-tiles of 16 (77 -> 80, masked below)
    v8f sc[5];
    #pragma unroll
    for (int t = 0; t < 5; ++t) sc[t] = (v8f){};
    #pragma unroll
    for (int t = 0; t < 5; ++t) {
        int s = t * 16 + nl; if (s >= S_) s = 0;
        const __bf16* kp = Kbf + (size_t)(b * S_ + s) * C_ + h * DH_;
        #pragma unroll
        for (int kt = 0; kt < 2; ++kt) {
            v16bf bK = loadB(kp + kt * 32, hf);
            sc[t] = __builtin_amdgcn_wmma_f32_16x16x32_bf16(
                false, aQ[kt], false, bK, (short)0, sc[t], false, false);
        }
    }
    // IP scores (T=16 exact)
    v8f sci = {};
    {
        const __bf16* kp = Kip + (size_t)(b * TIP + nl) * C_ + h * DH_;
        #pragma unroll
        for (int kt = 0; kt < 2; ++kt) {
            v16bf bK = loadB(kp + kt * 32, hf);
            sci = __builtin_amdgcn_wmma_f32_16x16x32_bf16(
                false, aQ[kt], false, bK, (short)0, sci, false, false);
        }
    }

    // softmax: row m = r + 8hf spans 16 lanes of this half x 5 tiles
    #pragma unroll
    for (int r = 0; r < 8; ++r) {
        const int mrow = r + (hf << 3);
        const int ml   = lt * 16 + mrow;
        float vals[5];
        #pragma unroll
        for (int t = 0; t < 5; ++t) {
            const int s = t * 16 + nl;
            float v = -1e30f;
            if (s < S_)
                v = sc[t][r] * scale + region[((size_t)b * L_ + ml) * S_ + s] * sigw;
            vals[t] = v;
        }
        float mx = vals[0];
        #pragma unroll
        for (int t = 1; t < 5; ++t) mx = fmaxf(mx, vals[t]);
        #pragma unroll
        for (int k = 1; k < 16; k <<= 1) mx = fmaxf(mx, __shfl_xor(mx, k, 32));
        float sum = 0.f;
        #pragma unroll
        for (int t = 0; t < 5; ++t) { vals[t] = __expf(vals[t] - mx); sum += vals[t]; }
        #pragma unroll
        for (int k = 1; k < 16; k <<= 1) sum += __shfl_xor(sum, k, 32);
        const float inv = 1.0f / sum;
        #pragma unroll
        for (int t = 0; t < 5; ++t)
            smem[wave][mrow * SP_ + t * 16 + nl] = f2bf(vals[t] * inv);
        smem[wave][mrow * SP_ + 80 + nl] = (__bf16)0.0f;     // pad 80..95

        float vi = sci[r] * scale;
        float mxi = vi;
        #pragma unroll
        for (int k = 1; k < 16; k <<= 1) mxi = fmaxf(mxi, __shfl_xor(mxi, k, 32));
        float pi = __expf(vi - mxi);
        float si = pi;
        #pragma unroll
        for (int k = 1; k < 16; k <<= 1) si += __shfl_xor(si, k, 32);
        smem[wave][PIP_ + mrow * TP_ + nl]      = f2bf(pi / si);
        smem[wave][PIP_ + mrow * TP_ + 16 + nl] = (__bf16)0.0f;  // pad k 16..31
    }
    __syncthreads();

    // out = P V + P_ip V_ip  (IP_SCALE = 1.0 -> same accumulators)
    v8f oacc[4];
    #pragma unroll
    for (int t4 = 0; t4 < 4; ++t4) oacc[t4] = (v8f){};

    #pragma unroll
    for (int kt = 0; kt < 3; ++kt) {                     // K = 96 padded S
        v16bf aP = loadA(&smem[wave][nl * SP_ + kt * 32], hf);
        #pragma unroll
        for (int t4 = 0; t4 < 4; ++t4) {
            const int col = h * DH_ + t4 * 16 + nl;
            v16bf bV = loadB(VT + ((size_t)b * C_ + col) * SP_ + kt * 32, hf);
            oacc[t4] = __builtin_amdgcn_wmma_f32_16x16x32_bf16(
                false, aP, false, bV, (short)0, oacc[t4], false, false);
        }
    }
    {
        v16bf aPi = loadA(&smem[wave][PIP_ + nl * TP_], hf);
        #pragma unroll
        for (int t4 = 0; t4 < 4; ++t4) {
            const int col = h * DH_ + t4 * 16 + nl;
            v16bf bV = loadB(VipT + ((size_t)b * C_ + col) * TP_, hf);
            oacc[t4] = __builtin_amdgcn_wmma_f32_16x16x32_bf16(
                false, aPi, false, bV, (short)0, oacc[t4], false, false);
        }
    }
    #pragma unroll
    for (int t4 = 0; t4 < 4; ++t4)
        #pragma unroll
        for (int r = 0; r < 8; ++r)
            attn_out[(size_t)(b * L_ + lt * 16 + r + (hf << 3)) * C_ +
                     h * DH_ + t4 * 16 + nl] = f2bf(oacc[t4][r]);
}

// ---------------------------------------------------------------------------
// Output GEMM (register-blocked 64x32, double-buffered):
//   f32 out = A @ WoutT^T + b_out + resid.   M = 8192 exact -> no guards.
// ---------------------------------------------------------------------------
__global__ __launch_bounds__(256) void out_gemm_blk(
    const __bf16* __restrict__ A, const __bf16* __restrict__ WT,
    const float* __restrict__ bias, const float* __restrict__ resid,
    float* __restrict__ out, int M, int N, int K, int tiles_n32)
{
    const int wave = threadIdx.x >> 5;
    const int lane = threadIdx.x & 31;
    const int wid  = blockIdx.x * 8 + wave;
    const int tm   = wid / tiles_n32;
    const int tn   = wid % tiles_n32;
    const int hf   = lane >> 4;
    const int nl   = lane & 15;

    const __bf16* ap[4];
    #pragma unroll
    for (int i = 0; i < 4; ++i)
        ap[i] = A + (size_t)(tm * 64 + i * 16 + nl) * K;
    const __bf16* bp[2];
    #pragma unroll
    for (int j = 0; j < 2; ++j)
        bp[j] = WT + (size_t)(tn * 32 + j * 16 + nl) * K;

    v8f acc[4][2] = {};
    v16bf a0[4], b0[2], a1[4], b1[2];
    #pragma unroll
    for (int i = 0; i < 4; ++i) a0[i] = loadA(ap[i], hf);
    #pragma unroll
    for (int j = 0; j < 2; ++j) b0[j] = loadB(bp[j], hf);

    for (int k0 = 0; k0 < K; k0 += 64) {
        #pragma unroll
        for (int i = 0; i < 4; ++i) a1[i] = loadA(ap[i] + k0 + 32, hf);
        #pragma unroll
        for (int j = 0; j < 2; ++j) b1[j] = loadB(bp[j] + k0 + 32, hf);
        #pragma unroll
        for (int i = 0; i < 4; ++i)
            #pragma unroll
            for (int j = 0; j < 2; ++j)
                acc[i][j] = __builtin_amdgcn_wmma_f32_16x16x32_bf16(
                    false, a0[i], false, b0[j], (short)0, acc[i][j], false, false);
        if (k0 + 64 < K) {
            #pragma unroll
            for (int i = 0; i < 4; ++i) a0[i] = loadA(ap[i] + k0 + 64, hf);
            #pragma unroll
            for (int j = 0; j < 2; ++j) b0[j] = loadB(bp[j] + k0 + 64, hf);
        }
        #pragma unroll
        for (int i = 0; i < 4; ++i)
            #pragma unroll
            for (int j = 0; j < 2; ++j)
                acc[i][j] = __builtin_amdgcn_wmma_f32_16x16x32_bf16(
                    false, a1[i], false, b1[j], (short)0, acc[i][j], false, false);
    }
    #pragma unroll
    for (int j = 0; j < 2; ++j) {
        const int ncol = tn * 32 + j * 16 + nl;
        const float bo = bias[ncol];
        #pragma unroll
        for (int i = 0; i < 4; ++i)
            #pragma unroll
            for (int r = 0; r < 8; ++r) {
                const size_t idx =
                    (size_t)(tm * 64 + i * 16 + r + (hf << 3)) * N + ncol;
                out[idx] = acc[i][j][r] + bo + resid[idx];
            }
    }
}

// ---------------------------------------------------------------------------
extern "C" void kernel_launch(void* const* d_in, const int* in_sizes, int n_in,
                              void* d_out, int out_size, void* d_ws, size_t ws_size,
                              hipStream_t stream) {
    const float* hs     = (const float*)d_in[0];   // [2,4096,1280]
    const float* enc    = (const float*)d_in[1];   // [2,77,768]
    const float* iph    = (const float*)d_in[2];   // [2,16,768]
    const float* region = (const float*)d_in[3];   // [2,4096,77]
    const float* sigma  = (const float*)d_in[4];   // [1]
    const float* Wq     = (const float*)d_in[5];
    const float* Wk     = (const float*)d_in[6];
    const float* Wv     = (const float*)d_in[7];
    const float* Wkip   = (const float*)d_in[8];
    const float* Wvip   = (const float*)d_in[9];
    const float* Wout   = (const float*)d_in[10];
    const float* bout   = (const float*)d_in[11];
    float* out = (float*)d_out;

    auto rnd = [](size_t x) { return (x + 255) & ~(size_t)255; };
    char* ws = (char*)d_ws;
    __bf16* hsb   = (__bf16*)ws; ws += rnd((size_t)B_ * L_ * C_ * 2);
    __bf16* Qbf   = (__bf16*)ws; ws += rnd((size_t)B_ * L_ * C_ * 2);
    __bf16* Obf   = (__bf16*)ws; ws += rnd((size_t)B_ * L_ * C_ * 2);
    __bf16* encb  = (__bf16*)ws; ws += rnd((size_t)B_ * S_ * CC_ * 2);
    __bf16* ipb   = (__bf16*)ws; ws += rnd((size_t)B_ * TIP * CC_ * 2);
    __bf16* WqT   = (__bf16*)ws; ws += rnd((size_t)C_ * C_ * 2);
    __bf16* WoT   = (__bf16*)ws; ws += rnd((size_t)C_ * C_ * 2);
    __bf16* WkT   = (__bf16*)ws; ws += rnd((size_t)C_ * CC_ * 2);
    __bf16* WvT   = (__bf16*)ws; ws += rnd((size_t)C_ * CC_ * 2);
    __bf16* WkiT  = (__bf16*)ws; ws += rnd((size_t)C_ * CC_ * 2);
    __bf16* WviT  = (__bf16*)ws; ws += rnd((size_t)C_ * CC_ * 2);
    __bf16* Kbf   = (__bf16*)ws; ws += rnd((size_t)B_ * S_ * C_ * 2);
    __bf16* Kipb  = (__bf16*)ws; ws += rnd((size_t)B_ * TIP * C_ * 2);
    __bf16* VT    = (__bf16*)ws; ws += rnd((size_t)B_ * C_ * SP_ * 2);
    __bf16* VipT  = (__bf16*)ws; ws += rnd((size_t)B_ * C_ * TP_ * 2);

    // zero padded V^T buffers (graph-capturable)
    hipMemsetAsync(VT,   0, (size_t)B_ * C_ * SP_ * 2, stream);
    hipMemsetAsync(VipT, 0, (size_t)B_ * C_ * TP_ * 2, stream);

    // activations -> bf16
    cvt_bf<<<5120, 256, 0, stream>>>(hs,  hsb,  B_ * L_ * C_ / 8);     // 1310720
    cvt_bf<<<58,   256, 0, stream>>>(enc, encb, B_ * S_ * CC_ / 8);    // 14784
    cvt_bf<<<12,   256, 0, stream>>>(iph, ipb,  B_ * TIP * CC_ / 8);   // 3072

    // weights -> transposed bf16 [N,K]
    transcvt<<<1600, 256, 0, stream>>>(Wq,   WqT,  C_,  C_);
    transcvt<<<1600, 256, 0, stream>>>(Wout, WoT,  C_,  C_);
    transcvt<<<960,  256, 0, stream>>>(Wk,   WkT,  CC_, C_);
    transcvt<<<960,  256, 0, stream>>>(Wv,   WvT,  CC_, C_);
    transcvt<<<960,  256, 0, stream>>>(Wkip, WkiT, CC_, C_);
    transcvt<<<960,  256, 0, stream>>>(Wvip, WviT, CC_, C_);

    const int tilesN32 = C_ / 32;                      // 40
    // Q: (8192/64)*(1280/32) = 128*40 = 5120 waves -> 640 blocks
    proj_gemm_blk<<<640, 256, 0, stream>>>(hsb, WqT, Qbf, B_ * L_, C_, C_, tilesN32);
    // K: ceil(154/64)=3 -> 3*40 = 120 waves -> 15 blocks
    proj_gemm_blk<<<15, 256, 0, stream>>>(encb, WkT, Kbf, B_ * S_, C_, CC_, tilesN32);
    // K_ip: 1*40 = 40 waves -> 5 blocks
    proj_gemm_blk<<<5, 256, 0, stream>>>(ipb, WkiT, Kipb, B_ * TIP, C_, CC_, tilesN32);
    // V / V_ip (transposed+padded store, single-tile kernel, tiles_n = 80)
    proj_gemm_T<<<100, 256, 0, stream>>>(encb, WvT, VT, B_ * S_, C_, CC_, C_ / 16, S_, SP_);
    proj_gemm_T<<<20, 256, 0, stream>>>(ipb, WviT, VipT, B_ * TIP, C_, CC_, C_ / 16, TIP, TP_);
    // fused attention: 2*20*256 = 10240 wave tiles -> 1280 blocks
    attn_fused<<<1280, 256, 0, stream>>>(Qbf, Kbf, VT, Kipb, VipT, region, sigma, Obf);
    // output projection + bias + residual: 5120 waves -> 640 blocks
    out_gemm_blk<<<640, 256, 0, stream>>>(Obf, WoT, bout, hs, out, B_ * L_, C_, C_, tilesN32);
}
